// GatedGIN_pyg_6133213298789
// MI455X (gfx1250) — compile-verified, matrix operand order checked by reading, not run on hardware
//
#include <hip/hip_runtime.h>

// ---------------------------------------------------------------------------
// GatedGIN for MI455X (gfx1250, wave32).
// Dense GEMMs: v_wmma_f32_16x16x32_bf16 (bf16 in, fp32 accumulate).
//   - block = 8 waves; all waves share one 16-col weight tile, each wave owns
//     one 16-row tile. Weight tile (4KB bf16) staged to LDS once per block via
//     GLOBAL_LOAD_ASYNC_TO_LDS_B128 (ASYNCcnt), B-fragments then read via DS.
// SpMM: coalesced float4 gathers + global fp32 atomics (L2-resident).
// Required ws_size ~ 598 MB.
// ---------------------------------------------------------------------------

#define NNODES 100000
#define NEDGES 1600000
#define HDIM   128
#define H3     384

typedef __attribute__((ext_vector_type(16))) __bf16 bf16x16;
typedef __attribute__((ext_vector_type(8)))  __bf16 bf16x8;
typedef __attribute__((ext_vector_type(8)))  float  f32x8;

__device__ __forceinline__ unsigned short f2bf(float f) {
  unsigned int u = __float_as_uint(f);
  u += 0x7FFFu + ((u >> 16) & 1u);   // round-to-nearest-even
  return (unsigned short)(u >> 16);
}

// ---- fp32 -> bf16 bits (grid-stride) --------------------------------------
__global__ void k_f32_to_bf16(const float* __restrict__ in,
                              unsigned short* __restrict__ out, int n) {
  for (int i = blockIdx.x * blockDim.x + threadIdx.x; i < n;
       i += gridDim.x * blockDim.x)
    out[i] = f2bf(in[i]);
}

__global__ void k_zero(float* __restrict__ p, int n) {
  for (int i = blockIdx.x * blockDim.x + threadIdx.x; i < n;
       i += gridDim.x * blockDim.x)
    p[i] = 0.0f;
}

// ---- SpMM: agg[dst] += w * x[src]; one wave per edge ----------------------
__global__ void k_spmm(const float* __restrict__ x, const int* __restrict__ dst,
                       const int* __restrict__ src, const float* __restrict__ ew,
                       float* __restrict__ agg) {
  int wid = (blockIdx.x * blockDim.x + threadIdx.x) >> 5;
  if (wid >= NEDGES) return;
  int lane = threadIdx.x & 31;
  int s = src[wid];
  int d = dst[wid];
  float w = ew[wid];
  const float4* xr = (const float4*)(x + (size_t)s * HDIM);
  float4 v = xr[lane];
  float* ar = agg + (size_t)d * HDIM + lane * 4;
  atomicAdd(ar + 0, w * v.x);
  atomicAdd(ar + 1, w * v.y);
  atomicAdd(ar + 2, w * v.z);
  atomicAdd(ar + 3, w * v.w);
}

// ---- WMMA A-fragment load: row-major [*,128] bf16 matrix in global --------
// 16-bit A/B layout (ISA 7.12.2): lane L holds row (L&15); K-chunks
// [kb, kb+8) and [kb+16, kb+24) with kb = (L&16)?8:0  -> two 16B loads.
__device__ __forceinline__ bf16x16 load_frag128(
    const unsigned short* __restrict__ base, int row, int k0, int lane) {
  const unsigned short* p =
      base + ((size_t)row << 7) + k0 + ((lane & 16) ? 8 : 0);
  union { bf16x16 v; bf16x8 h[2]; } u;
  u.h[0] = *(const bf16x8*)(p);
  u.h[1] = *(const bf16x8*)(p + 16);
  return u.v;
}

// ---- B-fragment load from the LDS-staged 16x128 weight tile ---------------
__device__ __forceinline__ bf16x16 load_frag_lds(
    const unsigned short* wl, int k0) {
  union { bf16x16 v; bf16x8 h[2]; } u;
  u.h[0] = *(const bf16x8*)(wl + k0);
  u.h[1] = *(const bf16x8*)(wl + k0 + 16);
  return u.v;
}

// ---- GEMM: C[N,M] = act(A[N,128] @ W[M,128]^T + bias) ---------------------
// grid = (ceil(rowTiles/8), M/16); block = 256 (8 waves).
// blockIdx.y selects the 16-row weight tile shared by all 8 waves.
template <int ACT>
__global__ void k_gemm_wmma(const unsigned short* __restrict__ A,
                            const unsigned short* __restrict__ W,
                            const float* __restrict__ bias,
                            float* __restrict__ C,
                            unsigned short* __restrict__ Cb,
                            int M, int rowTiles) {
  __shared__ unsigned short wtile[16 * 128];   // 4KB bf16 weight tile

  int tid = threadIdx.x;
  int wave = tid >> 5;
  int lane = tid & 31;
  int colBase = blockIdx.y << 4;

  // Stage weight tile to LDS: 256 threads x 16B = 4KB, via async DMA to LDS.
  {
    const unsigned short* src = W + (size_t)colBase * 128 + tid * 8;
    unsigned int ldsOff =
        (unsigned int)(size_t)(wtile + tid * 8);   // low 32b = LDS offset
    unsigned long long ga = (unsigned long long)(size_t)src;
    asm volatile("global_load_async_to_lds_b128 %0, %1, off"
                 :: "v"(ldsOff), "v"(ga) : "memory");
    asm volatile("s_wait_asynccnt 0x0" ::: "memory");
  }
  __syncthreads();

  int rowTile = blockIdx.x * 8 + wave;
  if (rowTile >= rowTiles) return;             // wave-uniform, after barrier

  int rowBase = rowTile << 4;
  int n = colBase + (lane & 15);               // output column of this lane
  float bv = bias[n];
  f32x8 c = {bv, bv, bv, bv, bv, bv, bv, bv};

  int arow = rowBase + (lane & 15);
  const unsigned short* wl =
      wtile + ((lane & 15) << 7) + ((lane & 16) ? 8 : 0);
#pragma unroll
  for (int k0 = 0; k0 < 128; k0 += 32) {
    bf16x16 a = load_frag128(A, arow, k0, lane);
    bf16x16 b = load_frag_lds(wl, k0);
    c = __builtin_amdgcn_wmma_f32_16x16x32_bf16(
        false, a, false, b, (short)0, c, false, false);
  }

  int rbase = rowBase + ((lane >> 4) << 3);    // rows {0..7} or {8..15}
#pragma unroll
  for (int j = 0; j < 8; ++j) {
    float v = c[j];
    if (ACT) v = v > 0.0f ? v : 0.0f;
    size_t off = (size_t)(rbase + j) * M + n;
    C[off] = v;
    if (Cb) Cb[off] = f2bf(v);
  }
}

// ---- GRU gates (elementwise) ----------------------------------------------
__global__ void k_gru(const float* __restrict__ gi, const float* __restrict__ gh,
                      const float* __restrict__ hprev,
                      unsigned short* __restrict__ hb) {
  int i = blockIdx.x * blockDim.x + threadIdx.x;
  if (i >= NNODES * HDIM) return;
  int node = i >> 7;
  int f = i & 127;
  const float* gir = gi + (size_t)node * H3;
  const float* ghr = gh + (size_t)node * H3;
  float ir = gir[f], iz = gir[128 + f], in_ = gir[256 + f];
  float hr = ghr[f], hz = ghr[128 + f], hn = ghr[256 + f];
  float r = 1.0f / (1.0f + __expf(-(ir + hr)));
  float z = 1.0f / (1.0f + __expf(-(iz + hz)));
  float nn = tanhf(in_ + r * hn);
  float hv = (1.0f - z) * nn + z * hprev[i];
  hb[i] = f2bf(hv);
}

// ---- row norm: x /= sqrt(1 + sum(x^2)); one wave per node -----------------
__global__ void k_norm(float* __restrict__ x, unsigned short* __restrict__ xb) {
  int wid = (blockIdx.x * blockDim.x + threadIdx.x) >> 5;
  if (wid >= NNODES) return;
  int lane = threadIdx.x & 31;
  float4* row = (float4*)(x + (size_t)wid * HDIM);
  float4 v = row[lane];
  float ss = v.x * v.x + v.y * v.y + v.z * v.z + v.w * v.w;
#pragma unroll
  for (int m = 16; m > 0; m >>= 1) ss += __shfl_xor(ss, m, 32);
  float inv = rsqrtf(1.0f + ss);
  v.x *= inv; v.y *= inv; v.z *= inv; v.w *= inv;
  row[lane] = v;
  unsigned short* rb = xb + (size_t)wid * HDIM + lane * 4;
  rb[0] = f2bf(v.x); rb[1] = f2bf(v.y); rb[2] = f2bf(v.z); rb[3] = f2bf(v.w);
}

// ---- small classifier GEMM: logits[N,M] = t[N,128] @ w[M,128]^T + b -------
__global__ void k_head2(const float* __restrict__ t, const float* __restrict__ w,
                        const float* __restrict__ b, float* __restrict__ logits,
                        int M) {
  int gid = blockIdx.x * blockDim.x + threadIdx.x;
  if (gid >= NNODES * M) return;
  int node = gid / M;
  int c = gid - node * M;
  const float* tr = t + (size_t)node * HDIM;
  const float* wr = w + (size_t)c * HDIM;
  float acc = b[c];
#pragma unroll 4
  for (int k = 0; k < HDIM; ++k) acc += tr[k] * wr[k];
  logits[(size_t)node * M + c] = acc;
}

// ---- log_softmax + softmax ------------------------------------------------
__global__ void k_softmax(const float* __restrict__ logits,
                          float* __restrict__ lsm, float* __restrict__ sm,
                          int M) {
  int node = blockIdx.x * blockDim.x + threadIdx.x;
  if (node >= NNODES) return;
  const float* lr = logits + (size_t)node * M;
  float mx = lr[0];
  for (int c = 1; c < M; ++c) mx = fmaxf(mx, lr[c]);
  float s = 0.0f;
  for (int c = 0; c < M; ++c) s += __expf(lr[c] - mx);
  float ls = __logf(s);
  for (int c = 0; c < M; ++c) {
    float v = lr[c] - mx - ls;
    lsm[(size_t)node * M + c] = v;
    sm[(size_t)node * M + c] = __expf(v);
  }
}

// ---------------------------------------------------------------------------
extern "C" void kernel_launch(void* const* d_in, const int* in_sizes, int n_in,
                              void* d_out, int out_size, void* d_ws,
                              size_t ws_size, hipStream_t stream) {
  const float* features    = (const float*)d_in[0];
  const int*   edge_index  = (const int*)d_in[1];
  const float* edge_weight = (const float*)d_in[2];
  const float* w1       = (const float*)d_in[3];
  const float* b1       = (const float*)d_in[4];
  const float* gin_w1   = (const float*)d_in[5];
  const float* gin_b1   = (const float*)d_in[6];
  const float* gin_w2   = (const float*)d_in[7];
  const float* gin_b2   = (const float*)d_in[8];
  const float* gru_wih  = (const float*)d_in[9];
  const float* gru_whh  = (const float*)d_in[10];
  const float* gru_bih  = (const float*)d_in[11];
  const float* gru_bhh  = (const float*)d_in[12];
  const float* head_w1  = (const float*)d_in[13];
  const float* head_b1  = (const float*)d_in[14];
  const float* head_w2[3] = {(const float*)d_in[15], (const float*)d_in[17],
                             (const float*)d_in[19]};
  const float* head_b2[3] = {(const float*)d_in[16], (const float*)d_in[18],
                             (const float*)d_in[20]};
  const int* dstp = edge_index;            // edge_index[0] = dst
  const int* srcp = edge_index + NEDGES;   // edge_index[1] = src
  float* out = (float*)d_out;

  // workspace layout
  char* wsb = (char*)d_ws;
  size_t off = 0;
  auto alloc = [&](size_t bytes) -> char* {
    char* p = wsb + off;
    off += (bytes + 255) & ~(size_t)255;
    return p;
  };
  const size_t NH  = (size_t)NNODES * HDIM;
  unsigned short* featb = (unsigned short*)alloc(NH * 2);
  unsigned short* xb    = (unsigned short*)alloc(NH * 2);
  unsigned short* aggb  = (unsigned short*)alloc(NH * 2);
  unsigned short* hb    = (unsigned short*)alloc(NH * 2);
  unsigned short* tb    = (unsigned short*)alloc(NH * 2);
  float* x32   = (float*)alloc(NH * 4);
  float* agg32 = (float*)alloc(NH * 4);
  float* t32   = (float*)alloc(NH * 4);
  float* gi32  = (float*)alloc((size_t)NNODES * H3 * 4);
  float* gh32  = (float*)alloc((size_t)NNODES * H3 * 4);
  float* logits = (float*)alloc((size_t)NNODES * 21 * 4);
  unsigned short* w1b  = (unsigned short*)alloc((size_t)HDIM * HDIM * 2);
  unsigned short* wihb = (unsigned short*)alloc((size_t)3 * H3 * HDIM * 2);
  unsigned short* whhb = (unsigned short*)alloc((size_t)3 * H3 * HDIM * 2);
  unsigned short* g1b  = (unsigned short*)alloc((size_t)3 * HDIM * HDIM * 2);
  unsigned short* g2b  = (unsigned short*)alloc((size_t)3 * HDIM * HDIM * 2);
  unsigned short* hw1b = (unsigned short*)alloc((size_t)3 * HDIM * HDIM * 2);

  const int TB = 256;
  auto cg = [](long n, int tb) { return (int)((n + tb - 1) / tb); };

  // per-call weight + feature conversion to bf16
  k_f32_to_bf16<<<cg(HDIM * HDIM, TB), TB, 0, stream>>>(w1, w1b, HDIM * HDIM);
  k_f32_to_bf16<<<cg(3 * H3 * HDIM, TB), TB, 0, stream>>>(gru_wih, wihb, 3 * H3 * HDIM);
  k_f32_to_bf16<<<cg(3 * H3 * HDIM, TB), TB, 0, stream>>>(gru_whh, whhb, 3 * H3 * HDIM);
  k_f32_to_bf16<<<cg(3 * HDIM * HDIM, TB), TB, 0, stream>>>(gin_w1, g1b, 3 * HDIM * HDIM);
  k_f32_to_bf16<<<cg(3 * HDIM * HDIM, TB), TB, 0, stream>>>(gin_w2, g2b, 3 * HDIM * HDIM);
  k_f32_to_bf16<<<cg(3 * HDIM * HDIM, TB), TB, 0, stream>>>(head_w1, hw1b, 3 * HDIM * HDIM);
  k_f32_to_bf16<<<cg((long)NH, TB), TB, 0, stream>>>(features, featb, (int)NH);

  const int rowTiles = NNODES / 16;                  // 6250
  const dim3 g128((rowTiles + 7) / 8, HDIM / 16);    // 782 x 8
  const dim3 g384((rowTiles + 7) / 8, H3 / 16);      // 782 x 24

  // mlp1: x = relu(features @ w1^T + b1)
  k_gemm_wmma<1><<<g128, 256, 0, stream>>>(featb, w1b, b1, x32, xb, HDIM,
                                           rowTiles);

  const size_t outLog[3] = {0, (size_t)NNODES * 2, (size_t)NNODES * 8};
  const size_t outSm0 = (size_t)NNODES * 29;
  const int Ms[3] = {2, 6, 21};

  for (int i = 0; i < 3; ++i) {
    // SpMM
    k_zero<<<cg((long)NH, TB), TB, 0, stream>>>(agg32, (int)NH);
    k_spmm<<<NEDGES / 8, 256, 0, stream>>>(x32, dstp, srcp, edge_weight, agg32);
    k_f32_to_bf16<<<cg((long)NH, TB), TB, 0, stream>>>(agg32, aggb, (int)NH);

    // GRU: gi = agg @ wih^T + bih ; gh = x @ whh^T + bhh ; gates
    k_gemm_wmma<0><<<g384, 256, 0, stream>>>(
        aggb, wihb + (size_t)i * H3 * HDIM, gru_bih + i * H3, gi32,
        (unsigned short*)nullptr, H3, rowTiles);
    k_gemm_wmma<0><<<g384, 256, 0, stream>>>(
        xb, whhb + (size_t)i * H3 * HDIM, gru_bhh + i * H3, gh32,
        (unsigned short*)nullptr, H3, rowTiles);
    k_gru<<<cg((long)NH, TB), TB, 0, stream>>>(gi32, gh32, x32, hb);

    // GIN MLP: t = relu(h @ g1^T + b1) ; x = relu(t @ g2^T + b2)
    k_gemm_wmma<1><<<g128, 256, 0, stream>>>(
        hb, g1b + (size_t)i * HDIM * HDIM, gin_b1 + i * HDIM, t32, tb, HDIM,
        rowTiles);
    k_gemm_wmma<1><<<g128, 256, 0, stream>>>(
        tb, g2b + (size_t)i * HDIM * HDIM, gin_b2 + i * HDIM, x32, xb, HDIM,
        rowTiles);

    // custom norm
    k_norm<<<NNODES / 8, 256, 0, stream>>>(x32, xb);

    // head: t = relu(x @ hw1^T + hb1) ; logits = t @ hw2^T + hb2
    k_gemm_wmma<1><<<g128, 256, 0, stream>>>(
        xb, hw1b + (size_t)i * HDIM * HDIM, head_b1 + i * HDIM, t32, tb, HDIM,
        rowTiles);
    k_head2<<<cg((long)NNODES * Ms[i], TB), TB, 0, stream>>>(
        t32, head_w2[i], head_b2[i], logits, Ms[i]);
    k_softmax<<<cg(NNODES, TB), TB, 0, stream>>>(
        logits, out + outLog[i], out + outSm0 + outLog[i], Ms[i]);
  }
  (void)in_sizes; (void)n_in; (void)out_size; (void)ws_size;
}
